// RingAttentionModel_4544075399707
// MI455X (gfx1250) — matrix operand order repaired and checked by reading, not compiled, not run on hardware
//
#include <hip/hip_runtime.h>
#include <hip/hip_bf16.h>
#include <cstdint>

#define D_MODEL 2048
#define NHEADS  16
#define HEAD_DIM 128
#define BATCH   2
#define SEQ     2048
#define M_TOT   (BATCH * SEQ)   // 4096 rows for all projection GEMMs

typedef __attribute__((ext_vector_type(16))) _Float16 v16h;
typedef __attribute__((ext_vector_type(8)))  _Float16 v8h;
typedef __attribute__((ext_vector_type(8)))  float    v8f;

union F16Frag { v16h v; v8h h[2]; };

__device__ __forceinline__ v8f wmma_f16(v16h a, v16h b, v8f c) {
  // D = A(16x32 f16) * B(32x16 f16) + C(16x16 f32)
  return __builtin_amdgcn_wmma_f32_16x16x32_f16(false, a, false, b, (short)0, c,
                                                false, false);
}

// Async copy: 16B global -> LDS per lane, tracked by ASYNCcnt (no VGPR data).
__device__ __forceinline__ void async_load_b128(uint32_t lds_off, const void* gaddr) {
  asm volatile("global_load_async_to_lds_b128 %0, %1, off"
               :: "v"(lds_off), "v"((uint64_t)(uintptr_t)gaddr)
               : "memory");
}

// ---- Fragment loaders (layout assumptions localized here) ----
// A fragment 16x32 (MxK), f16, source row-major [m][k].
// ISA 7.12.2: lane m=l&15, kg=l>>4; halves j: K = (j>>3)*16 + kg*8 + (j&7)
__device__ __forceinline__ v16h load_a16x32(const _Float16* src, int ld,
                                            int m0, int k0, int lane) {
  int m  = m0 + (lane & 15);
  int kg = (lane >> 4) & 1;
  const _Float16* p = src + (size_t)m * ld + k0 + kg * 8;
  F16Frag f;
  f.h[0] = *(const v8h*)(p);        // K = kg*8 + 0..7
  f.h[1] = *(const v8h*)(p + 16);   // K = 16 + kg*8 + 0..7
  return f.v;
}

// B fragment 32x16 (KxN), f16, with B[kk][n] = src[n0+n][k0+kk] (src row-major).
// halves j: K = (j>>3)*16 + kg*8 + (j&7), same as A fragment.
__device__ __forceinline__ v16h load_b32x16(const _Float16* src, int ld,
                                            int n0, int k0, int lane) {
  int n  = n0 + (lane & 15);
  int kg = (lane >> 4) & 1;
  const _Float16* p = src + (size_t)n * ld + k0 + kg * 8;
  F16Frag f;
  f.h[0] = *(const v8h*)(p);
  f.h[1] = *(const v8h*)(p + 16);
  return f.v;
}

// ---- fp32 -> f16 convert ----
__global__ void cvt_f32_to_f16(const float* __restrict__ src,
                               _Float16* __restrict__ dst, int n) {
  int i = blockIdx.x * blockDim.x + threadIdx.x;
  if (i < n) dst[i] = (_Float16)src[i];
}

// ---- V[B,S,H,Dh] -> Vt[B,H,Dh,S] ----
__global__ void transpose_v_kernel(const _Float16* __restrict__ V,
                                   _Float16* __restrict__ Vt) {
  size_t i = (size_t)blockIdx.x * blockDim.x + threadIdx.x;
  int d = (int)(i & (HEAD_DIM - 1));
  int h = (int)((i >> 7) & (NHEADS - 1));
  int s = (int)((i >> 11) & (SEQ - 1));
  int b = (int)(i >> 22);
  Vt[((size_t)(b * NHEADS + h) * HEAD_DIM + d) * SEQ + s] = V[i];
}

// ---- Y[M,N] = X[M,K] * W[N,K]^T  (M=4096, N=K=2048) ----
// Block 256 threads = 8 waves (4x2); wave tile 32x64; block tile 128x128.
// Double-buffered LDS staging via global_load_async_to_lds_b128, K-step 64.
#define KSTEP   64
#define LDS_LDA 72   // 64 halves + 8 pad halves (16B) -> conflict-free frag reads

template <bool STORE_F32>
__global__ __launch_bounds__(256) void gemm_xwt_kernel(
    const _Float16* __restrict__ X, const _Float16* __restrict__ W,
    _Float16* __restrict__ Yh, float* __restrict__ Yf) {
  const int K = D_MODEL, N = D_MODEL;
  __shared__ __align__(128) _Float16 Ah[2][128 * LDS_LDA];
  __shared__ __align__(128) _Float16 Bh[2][128 * LDS_LDA];

  int tid  = threadIdx.x;
  int lane = tid & 31;
  int wave = tid >> 5;
  int wm = wave >> 1, wn = wave & 1;
  int m0b = blockIdx.x * 128;
  int n0b = blockIdx.y * 128;

  // Stage one 128x64 A tile + 128x64 B tile into buffer `buf` (async).
  // 2KB per tile per wave-issue: 8 async b128 instructions per thread/stage.
  auto stage = [&](int buf, int k0) {
#pragma unroll
    for (int i = 0; i < 4; ++i) {
      int c   = tid + i * 256;     // 0..1023 : 128 rows x 8 16B segments
      int row = c >> 3;
      int seg = c & 7;
      const _Float16* ga = X + (size_t)(m0b + row) * K + k0 + seg * 8;
      uint32_t la = (uint32_t)(uintptr_t)&Ah[buf][row * LDS_LDA + seg * 8];
      async_load_b128(la, ga);
      const _Float16* gb = W + (size_t)(n0b + row) * K + k0 + seg * 8;
      uint32_t lb = (uint32_t)(uintptr_t)&Bh[buf][row * LDS_LDA + seg * 8];
      async_load_b128(lb, gb);
    }
  };

  const v8f vzero = {};
  v8f acc[2][4];
#pragma unroll
  for (int i = 0; i < 2; ++i)
#pragma unroll
    for (int j = 0; j < 4; ++j) acc[i][j] = vzero;

  stage(0, 0);
  stage(1, KSTEP);

  for (int k0 = 0, it = 0; k0 < K; k0 += KSTEP, ++it) {
    int cur = it & 1;
    // stage `it` complete when only stage `it+1`'s 8 asyncs remain
    if (k0 + KSTEP < K) asm volatile("s_wait_asynccnt 0x8" ::: "memory");
    else                asm volatile("s_wait_asynccnt 0x0" ::: "memory");
    __syncthreads();
    const _Float16* At = &Ah[cur][0];
    const _Float16* Bt = &Bh[cur][0];
#pragma unroll
    for (int kc = 0; kc < 2; ++kc) {
      v16h a0 = load_a16x32(At, LDS_LDA, wm * 32,      kc * 32, lane);
      v16h a1 = load_a16x32(At, LDS_LDA, wm * 32 + 16, kc * 32, lane);
#pragma unroll
      for (int j = 0; j < 4; ++j) {
        v16h b = load_b32x16(Bt, LDS_LDA, wn * 64 + 16 * j, kc * 32, lane);
        acc[0][j] = wmma_f16(a0, b, acc[0][j]);
        acc[1][j] = wmma_f16(a1, b, acc[1][j]);
      }
    }
    __syncthreads();
    if (k0 + 2 * KSTEP < K) stage(cur, k0 + 2 * KSTEP);
  }

  int nn = lane & 15, mg = (lane >> 4) * 8;
  int m0 = m0b + wm * 32, n0 = n0b + wn * 64;
#pragma unroll
  for (int i = 0; i < 2; ++i)
#pragma unroll
    for (int j = 0; j < 4; ++j)
#pragma unroll
      for (int r = 0; r < 8; ++r) {
        size_t idx = (size_t)(m0 + 16 * i + mg + r) * N + (n0 + 16 * j + nn);
        if constexpr (STORE_F32) Yf[idx] = acc[i][j][r];
        else                     Yh[idx] = (_Float16)acc[i][j][r];
      }
}

// ---- Causal flash attention: one wave per (b, h, 16-query block) ----
// Q,K: [B,S,H*Dh] f16; Vt: [B,H,Dh,S] f16; O: [B,S,H*Dh] f16
__global__ __launch_bounds__(256) void flash_attn_kernel(
    const _Float16* __restrict__ Q, const _Float16* __restrict__ Kh,
    const _Float16* __restrict__ Vt, _Float16* __restrict__ O) {
  __shared__ __align__(64) _Float16 lds_p[8][16 * 32];  // P tile per wave
  int lane = threadIdx.x & 31;
  int wave = threadIdx.x >> 5;
  int gw = blockIdx.x * 8 + wave;         // 0..4095
  int qb = gw & (SEQ / 16 - 1);           // 128 query blocks
  int h  = (gw >> 7) & (NHEADS - 1);
  int b  = gw >> 11;
  int q0 = qb * 16;

  const _Float16* qhead  = Q  + ((size_t)b * SEQ) * D_MODEL + h * HEAD_DIM;
  const _Float16* khead  = Kh + ((size_t)b * SEQ) * D_MODEL + h * HEAD_DIM;
  const _Float16* vthead = Vt + (size_t)(b * NHEADS + h) * HEAD_DIM * SEQ;
  _Float16* lp = &lds_p[wave][0];

  v16h qf[4];
#pragma unroll
  for (int kc = 0; kc < 4; ++kc)
    qf[kc] = load_a16x32(qhead, D_MODEL, q0, kc * 32, lane);

  const v8f vzero = {};
  v8f o[8];
  float m_i[8], l_i[8];
#pragma unroll
  for (int r = 0; r < 8; ++r) { m_i[r] = -3.0e30f; l_i[r] = 0.0f; }
#pragma unroll
  for (int dd = 0; dd < 8; ++dd) o[dd] = vzero;

  const float sm_scale = 0.088388347648318447f;  // 1/sqrt(128)
  int nn = lane & 15;
  int mg = (lane >> 4) * 8;
  int npair = ((q0 + 15) >> 5) + 1;  // 32-key blocks in the causal range

  for (int jp = 0; jp < npair; ++jp) {
    int k0 = jp * 32;
    v8f s0 = vzero, s1 = vzero;
#pragma unroll
    for (int kc = 0; kc < 4; ++kc) {
      v16h kb0 = load_b32x16(khead, D_MODEL, k0,      kc * 32, lane);
      v16h kb1 = load_b32x16(khead, D_MODEL, k0 + 16, kc * 32, lane);
      s0 = wmma_f16(qf[kc], kb0, s0);
      s1 = wmma_f16(qf[kc], kb1, s1);
    }
    float alpha[8];
#pragma unroll
    for (int r = 0; r < 8; ++r) {
      int qrow = q0 + mg + r;
      float e0 = s0[r] * sm_scale;
      float e1 = s1[r] * sm_scale;
      if (k0 + nn > qrow)      e0 = -3.0e30f;   // causal mask
      if (k0 + 16 + nn > qrow) e1 = -3.0e30f;
      float mx = fmaxf(e0, e1);
      mx = fmaxf(mx, __shfl_xor(mx, 1));
      mx = fmaxf(mx, __shfl_xor(mx, 2));
      mx = fmaxf(mx, __shfl_xor(mx, 4));
      mx = fmaxf(mx, __shfl_xor(mx, 8));
      float mnew = fmaxf(m_i[r], mx);
      float a  = __expf(m_i[r] - mnew);
      float p0 = __expf(e0 - mnew);
      float p1 = __expf(e1 - mnew);
      float rs = p0 + p1;
      rs += __shfl_xor(rs, 1);
      rs += __shfl_xor(rs, 2);
      rs += __shfl_xor(rs, 4);
      rs += __shfl_xor(rs, 8);
      l_i[r] = l_i[r] * a + rs;
      m_i[r] = mnew;
      alpha[r] = a;
      // stage P (C-layout) into LDS row-major [m][kk] for A-layout re-read
      lp[(mg + r) * 32 + nn]      = (_Float16)p0;
      lp[(mg + r) * 32 + 16 + nn] = (_Float16)p1;
    }
#pragma unroll
    for (int dd = 0; dd < 8; ++dd)
#pragma unroll
      for (int r = 0; r < 8; ++r)
        o[dd][r] = o[dd][r] * alpha[r];

    // cross-lane LDS visibility within the wave before reloading P
    asm volatile("s_wait_dscnt 0" ::: "memory");
    v16h pf = load_a16x32(lp, 32, 0, 0, lane);
#pragma unroll
    for (int dd = 0; dd < 8; ++dd) {
      v16h vb = load_b32x16(vthead, SEQ, dd * 16, k0, lane);
      o[dd] = wmma_f16(pf, vb, o[dd]);   // O += P(16x32) * V(32x16)
    }
  }

  float inv[8];
#pragma unroll
  for (int r = 0; r < 8; ++r) inv[r] = 1.0f / l_i[r];
#pragma unroll
  for (int dd = 0; dd < 8; ++dd)
#pragma unroll
    for (int r = 0; r < 8; ++r) {
      size_t idx = (size_t)(b * SEQ + q0 + mg + r) * D_MODEL +
                   h * HEAD_DIM + dd * 16 + nn;
      O[idx] = (_Float16)(o[dd][r] * inv[r]);
    }
}

extern "C" void kernel_launch(void* const* d_in, const int* in_sizes, int n_in,
                              void* d_out, int out_size, void* d_ws, size_t ws_size,
                              hipStream_t stream) {
  (void)in_sizes; (void)n_in; (void)out_size; (void)ws_size;
  const float* x  = (const float*)d_in[0];
  const float* Wq = (const float*)d_in[1];
  const float* Wk = (const float*)d_in[2];
  const float* Wv = (const float*)d_in[3];
  const float* Wo = (const float*)d_in[4];
  float* out = (float*)d_out;

  const size_t nX = (size_t)M_TOT * D_MODEL;    // 8388608
  const size_t nW = (size_t)D_MODEL * D_MODEL;  // 4194304

  _Float16* ws  = (_Float16*)d_ws;
  _Float16* xh  = ws; ws += nX;
  _Float16* wqh = ws; ws += nW;
  _Float16* wkh = ws; ws += nW;
  _Float16* wvh = ws; ws += nW;
  _Float16* woh = ws; ws += nW;
  _Float16* qh  = ws; ws += nX;
  _Float16* kh  = ws; ws += nX;
  _Float16* vh  = ws; ws += nX;
  _Float16* vth = ws; ws += nX;
  _Float16* ah  = xh;   // x no longer needed after projections; reuse for attn out

  dim3 blk(256);
  cvt_f32_to_f16<<<dim3((unsigned)((nX + 255) / 256)), blk, 0, stream>>>(x,  xh,  (int)nX);
  cvt_f32_to_f16<<<dim3((unsigned)((nW + 255) / 256)), blk, 0, stream>>>(Wq, wqh, (int)nW);
  cvt_f32_to_f16<<<dim3((unsigned)((nW + 255) / 256)), blk, 0, stream>>>(Wk, wkh, (int)nW);
  cvt_f32_to_f16<<<dim3((unsigned)((nW + 255) / 256)), blk, 0, stream>>>(Wv, wvh, (int)nW);
  cvt_f32_to_f16<<<dim3((unsigned)((nW + 255) / 256)), blk, 0, stream>>>(Wo, woh, (int)nW);

  dim3 ggrid(M_TOT / 128, D_MODEL / 128);  // 32 x 16
  gemm_xwt_kernel<false><<<ggrid, blk, 0, stream>>>(xh, wqh, qh, nullptr);
  gemm_xwt_kernel<false><<<ggrid, blk, 0, stream>>>(xh, wkh, kh, nullptr);
  gemm_xwt_kernel<false><<<ggrid, blk, 0, stream>>>(xh, wvh, vh, nullptr);

  transpose_v_kernel<<<dim3((unsigned)(nX / 256)), blk, 0, stream>>>(vh, vth);

  flash_attn_kernel<<<dim3(BATCH * NHEADS * (SEQ / 16) / 8), blk, 0, stream>>>(
      qh, kh, vth, ah);

  gemm_xwt_kernel<true><<<ggrid, blk, 0, stream>>>(ah, woh, nullptr, out);
}
